// GKATMultiHead_27556510171820
// MI455X (gfx1250) — matrix operand order repaired and checked by reading, not compiled, not run on hardware
//
#include <hip/hip_runtime.h>
#include <hip/hip_bf16.h>

// GKAT multi-head: out = elu( (P @ (ca@v)) / (P @ rowsum(ca) + eps) ),  P = exp(q k^T / 8)
// Row-max subtraction is mathematically a no-op for the normalized output, and the
// per-row normalizer commutes out of the final matmul -> no T x T x T GEMM needed.
//
// Pipeline (all f16 WMMA, fp32 accum):
//   0) cvt:    feat/ca/W  f32 -> f16
//   1) qkv:    GEMM feat(4096x512) x W^T -> Q,K (B,H,T,64) row-major, V^T (B,H,64,T); Q pre-scaled by 1/8
//   2) rowsum: s[b][t] = sum_j ca[b][t][j]            (f32)
//   3) fill:   UT rows 64..79 = [s ; zeros]           (f16)
//   4) u:      UT[bh][0..63][t] = (ca @ v)^T          (WMMA GEMM, transposed store)
//   5) flash:  per 16-row tile: S = Q K^T (WMMA), P = exp(S) -> LDS -> A-frag,
//              acc(16x80) += P @ UT-cols (WMMA), out = elu(num / (den + eps))
//
// Workspace: ~40.4 MB.

typedef __attribute__((ext_vector_type(16))) _Float16 v16h;
typedef __attribute__((ext_vector_type(8)))  _Float16 v8h;
typedef __attribute__((ext_vector_type(4)))  _Float16 v4h;
typedef __attribute__((ext_vector_type(8)))  float    v8f;

#define B_      2
#define T_      2048
#define INDIM_  512
#define H_      8
#define D_      64
#define BH_     (B_ * H_)
#define UROWS_  80
#define SCALE_  0.125f
#define EPS_    1e-9f

static __device__ __forceinline__ v8f wmma_f16(v16h a, v16h b, v8f c) {
  return __builtin_amdgcn_wmma_f32_16x16x32_f16(false, a, false, b, (short)0, c,
                                                false, false);
}

// A-operand fragment, f16 16x32, from row-major tile (row stride `ld` halves).
// Lanes 0..15: row=lane, K {0..15}; lanes 16..31: row=lane-16, K {16..31}.
static __device__ __forceinline__ v16h frag_a(const _Float16* tile, int ld, int lane) {
  int r  = lane & 15;
  int ko = (lane & 16) ? 16 : 0;
  const _Float16* p = tile + (long)r * ld + ko;
  v8h lo = *(const v8h*)p;         // K = ko .. ko+7       -> VGPR0..3
  v8h hi = *(const v8h*)(p + 8);   // K = ko+8 .. ko+15    -> VGPR4..7
  v16h a;
#pragma unroll
  for (int i = 0; i < 8; ++i) { a[i] = lo[i]; a[i + 8] = hi[i]; }
  return a;
}

// ---------------- kernel 0: f32 -> f16 conversion (feat, ca, W) ----------------
__global__ void __launch_bounds__(256)
cvt_kernel(const float* feat, const float* ca, const float* w,
           _Float16* feat16, _Float16* ca16, _Float16* w16) {
  const long FEAT_N = (long)B_ * T_ * INDIM_;   // 2,097,152
  const long CA_N   = (long)B_ * T_ * T_;       // 8,388,608
  const long W_N    = (long)3 * 512 * INDIM_;   //   786,432
  long i4 = ((long)blockIdx.x * blockDim.x + threadIdx.x) * 4;
  const float* src; _Float16* dst; long off;
  if (i4 < FEAT_N)                  { src = feat; dst = feat16; off = i4; }
  else if (i4 < FEAT_N + CA_N)      { src = ca;   dst = ca16;   off = i4 - FEAT_N; }
  else if (i4 < FEAT_N + CA_N+W_N)  { src = w;    dst = w16;    off = i4 - FEAT_N - CA_N; }
  else return;
  float4 v = *(const float4*)(src + off);
  v4h h; h[0] = (_Float16)v.x; h[1] = (_Float16)v.y; h[2] = (_Float16)v.z; h[3] = (_Float16)v.w;
  *(v4h*)(dst + off) = h;
}

// ---------------- kernel 1: QKV projection GEMM ----------------
// qkv col o = d*24 + s*8 + h. Each wave: one 16(M) x 16(d) tile for fixed (s,h).
__global__ void __launch_bounds__(128)
qkv_kernel(const _Float16* feat16, const _Float16* w16,
           _Float16* q16, _Float16* k16, _Float16* vt16) {
  int lane = threadIdx.x & 31;
  long tile = (long)blockIdx.x * 4 + (threadIdx.x >> 5);  // 256 * 96 tiles
  int  ncode = (int)(tile % 96);
  long mtile = tile / 96;
  int dt = ncode & 3, sh = ncode >> 2, h = sh & 7, s = sh >> 3;
  long row0 = mtile * 16;                                  // flat row over B*T

  const _Float16* arow = feat16 + row0 * INDIM_;
  int bc = lane & 15, bko = (lane & 16) ? 16 : 0;
  // B column n -> W row (dt*16+n)*24 + s*8 + h (rows are K-contiguous)
  const _Float16* bcol = w16 + ((long)((dt * 16 + bc) * 24 + s * 8 + h)) * INDIM_ + bko;

  v8f acc = {};
#pragma unroll 4
  for (int k = 0; k < INDIM_; k += 32) {
    v16h a = frag_a(arow + k, INDIM_, lane);
    v16h b = *(const v16h*)(bcol + k);
    acc = wmma_f16(a, b, acc);
  }

  int col   = lane & 15;
  int rbase = (lane & 16) ? 8 : 0;
  int d     = dt * 16 + col;
  int bb    = (int)(row0 / T_);
  int t0    = (int)(row0 % T_);
  if (s == 2) {                 // V^T[b][h][d][t] : 8 consecutive t -> one b128 store
    v8h hv;
#pragma unroll
    for (int c = 0; c < 8; ++c) hv[c] = (_Float16)acc[c];
    _Float16* dst = vt16 + (((long)(bb * H_ + h) * D_ + d) * T_) + t0 + rbase;
    *(v8h*)dst = hv;
  } else {                      // Q/K row-major (B,H,T,D); Q pre-scaled by 1/8
    float sc = (s == 0) ? SCALE_ : 1.0f;
    _Float16* dst = (s == 0 ? q16 : k16) +
                    ((long)(bb * H_ + h) * T_ + t0 + rbase) * D_ + d;
#pragma unroll
    for (int c = 0; c < 8; ++c) dst[(long)c * D_] = (_Float16)(acc[c] * sc);
  }
}

// ---------------- kernel 2: s[b][t] = rowsum(ca) ----------------
__global__ void __launch_bounds__(256)
rowsum_kernel(const float* ca, float* s32) {
  int lane = threadIdx.x & 31;
  long row = (long)blockIdx.x * 8 + (threadIdx.x >> 5);   // 4096 rows
  const float* p = ca + row * T_;
  float acc = 0.f;
  for (int j = lane * 4; j < T_; j += 128) {
    float4 v = *(const float4*)(p + j);
    acc += v.x + v.y + v.z + v.w;
  }
#pragma unroll
  for (int off = 16; off; off >>= 1) acc += __shfl_xor(acc, off, 32);
  if (lane == 0) s32[row] = acc;
}

// ---------------- kernel 3: UT rows 64..79 = [s ; 0] ----------------
__global__ void __launch_bounds__(256)
fill_kernel(const float* s32, _Float16* ut16) {
  long idx = (long)blockIdx.x * 256 + threadIdx.x;        // BH*16*2048 = 1<<19
  int t  = (int)(idx & (T_ - 1));
  int r  = (int)((idx >> 11) & 15);
  int bh = (int)(idx >> 15);
  int b  = bh >> 3;
  _Float16 val = (r == 0) ? (_Float16)s32[(long)b * T_ + t] : (_Float16)0.0f;
  ut16[((long)bh * UROWS_ + 64 + r) * T_ + t] = val;
}

// ---------------- kernel 4: UT[bh][0..63][t] = (ca @ v)^T ----------------
__global__ void __launch_bounds__(128)
u_kernel(const _Float16* ca16, const _Float16* vt16, _Float16* ut16) {
  int lane = threadIdx.x & 31;
  long tile = (long)blockIdx.x * 4 + (threadIdx.x >> 5);  // BH*128*4 = 8192
  int nt = (int)(tile & 3);
  int mt = (int)((tile >> 2) & 127);
  int bh = (int)(tile >> 9);
  int b  = bh >> 3;
  int row0 = mt * 16, d0 = nt * 16;

  const _Float16* arow = ca16 + ((long)b * T_ + row0) * T_;
  int bc = lane & 15, bko = (lane & 16) ? 16 : 0;
  const _Float16* bcol = vt16 + ((long)bh * D_ + d0 + bc) * T_ + bko;

  v8f acc = {};
  for (int k = 0; k < T_; k += 32) {
    v16h a = frag_a(arow + k, T_, lane);
    v16h bf = *(const v16h*)(bcol + k);
    acc = wmma_f16(a, bf, acc);
  }
  // transposed store: fixed d = d0+(lane&15), 8 consecutive t -> one b128
  int rbase = (lane & 16) ? 8 : 0;
  v8h hv;
#pragma unroll
  for (int c = 0; c < 8; ++c) hv[c] = (_Float16)acc[c];
  _Float16* dst = ut16 + ((long)bh * UROWS_ + d0 + (lane & 15)) * T_ + row0 + rbase;
  *(v8h*)dst = hv;
}

// ---------------- kernel 5: flash pass ----------------
__global__ void __launch_bounds__(128)
flash_kernel(const _Float16* q16, const _Float16* k16, const _Float16* ut16,
             float* out) {
  __shared__ _Float16 lds[4][16 * 40];   // per-wave 16x32 P tile, row stride 40
  int lane = threadIdx.x & 31;
  int wav  = threadIdx.x >> 5;
  long tile = (long)blockIdx.x * 4 + wav;   // BH*128 = 2048 row-tiles
  int mt = (int)(tile & 127);
  int bh = (int)(tile >> 7);
  int b = bh >> 3, h = bh & 7;
  int row0 = mt * 16;

  const _Float16* qrow = q16 + ((long)bh * T_ + row0) * D_;
  v16h qa0 = frag_a(qrow, D_, lane);        // K = 0..31
  v16h qa1 = frag_a(qrow + 32, D_, lane);   // K = 32..63

  int bc = lane & 15, bko = (lane & 16) ? 16 : 0;
  const _Float16* kbase = k16 + (long)bh * T_ * D_;
  const _Float16* ubase = ut16 + (long)bh * UROWS_ * T_;

  v8f acc0 = {}, acc1 = {}, acc2 = {}, acc3 = {}, acc4 = {};
  _Float16* myl = &lds[wav][0];
  int ccol = lane & 15, crb = (lane & 16) ? 8 : 0;

  for (int j = 0; j < T_; j += 32) {
    // ---- S = Q K^T for columns j..j+31 (two 16x16 tiles, K-dim 64) ----
    const _Float16* kc0 = kbase + (long)(j + bc) * D_ + bko;
    const _Float16* kc1 = kbase + (long)(j + 16 + bc) * D_ + bko;
    __builtin_prefetch(kbase + (long)(j + 32 + bc) * D_, 0, 3);
    v16h kb00 = *(const v16h*)(kc0);
    v16h kb01 = *(const v16h*)(kc0 + 32);
    v16h kb10 = *(const v16h*)(kc1);
    v16h kb11 = *(const v16h*)(kc1 + 32);
    v8f s0 = {}, s1 = {};
    s0 = wmma_f16(qa0, kb00, s0);
    s0 = wmma_f16(qa1, kb01, s0);
    s1 = wmma_f16(qa0, kb10, s1);
    s1 = wmma_f16(qa1, kb11, s1);

    // ---- P = exp(S) (row-shift is a mathematical no-op here), stage to LDS ----
#pragma unroll
    for (int c = 0; c < 8; ++c) {
      myl[(crb + c) * 40 + ccol]      = (_Float16)__expf(s0[c]);
      myl[(crb + c) * 40 + 16 + ccol] = (_Float16)__expf(s1[c]);
    }
    asm volatile("s_wait_dscnt 0" ::: "memory");   // same-wave DS RAW

    // ---- reload P as 16x32 A fragment ----
    {
      int r = lane & 15, ko = (lane & 16) ? 16 : 0;
      const _Float16* pp = myl + r * 40 + ko;
      v16h pa;
#pragma unroll
      for (int i = 0; i < 8; ++i) { pa[i] = pp[i]; pa[i + 8] = pp[8 + i]; }

      // ---- acc(16x80) += P @ UT columns (5 N-tiles: 64 dims + denom) ----
      const _Float16* uc = ubase + (long)bc * T_ + j + bko;
      acc0 = wmma_f16(pa, *(const v16h*)(uc + (long)0  * 16 * T_), acc0);
      acc1 = wmma_f16(pa, *(const v16h*)(uc + (long)1  * 16 * T_), acc1);
      acc2 = wmma_f16(pa, *(const v16h*)(uc + (long)2  * 16 * T_), acc2);
      acc3 = wmma_f16(pa, *(const v16h*)(uc + (long)3  * 16 * T_), acc3);
      acc4 = wmma_f16(pa, *(const v16h*)(uc + (long)4  * 16 * T_), acc4);
    }
  }

  // ---- normalize (denominator = column 64, lives in lanes 0 / 16), elu, store ----
  int srcl = lane & 16;
#pragma unroll
  for (int c = 0; c < 8; ++c) {
    float den = __shfl(acc4[c], srcl, 32) + EPS_;
    float inv = 1.0f / den;
    int t = row0 + crb + c;
    float* orow = out + ((long)b * T_ + t) * (H_ * D_) + h * D_ + ccol;
    float x;
    x = acc0[c] * inv; orow[0]  = x > 0.f ? x : __expf(x) - 1.0f;
    x = acc1[c] * inv; orow[16] = x > 0.f ? x : __expf(x) - 1.0f;
    x = acc2[c] * inv; orow[32] = x > 0.f ? x : __expf(x) - 1.0f;
    x = acc3[c] * inv; orow[48] = x > 0.f ? x : __expf(x) - 1.0f;
  }
}

// ---------------- launch ----------------
extern "C" void kernel_launch(void* const* d_in, const int* in_sizes, int n_in,
                              void* d_out, int out_size, void* d_ws, size_t ws_size,
                              hipStream_t stream) {
  const float* feat = (const float*)d_in[0];   // (2, 2048, 512)  f32
  const float* ca   = (const float*)d_in[1];   // (2, 2048, 2048) f32
  const float* w    = (const float*)d_in[2];   // (1536, 512)     f32
  float* out = (float*)d_out;                  // (2, 2048, 512)  f32

  char* ws = (char*)d_ws;
  const size_t FEAT16 = (size_t)B_ * T_ * INDIM_ * 2;          // 4 MB
  const size_t W16    = (size_t)3 * 512 * INDIM_ * 2;          // 1.5 MB
  const size_t CA16   = (size_t)B_ * T_ * T_ * 2;              // 16 MB
  const size_t QK16   = (size_t)BH_ * T_ * D_ * 2;             // 4 MB each
  const size_t UT16   = (size_t)BH_ * UROWS_ * T_ * 2;         // 5 MB

  _Float16* feat16 = (_Float16*)(ws);
  _Float16* w16    = (_Float16*)(ws + FEAT16);
  _Float16* ca16   = (_Float16*)(ws + FEAT16 + W16);
  _Float16* q16    = (_Float16*)(ws + FEAT16 + W16 + CA16);
  _Float16* k16    = (_Float16*)(ws + FEAT16 + W16 + CA16 + QK16);
  _Float16* vt16   = (_Float16*)(ws + FEAT16 + W16 + CA16 + 2 * QK16);
  _Float16* ut16   = (_Float16*)(ws + FEAT16 + W16 + CA16 + 3 * QK16);
  float*    s32    = (float*)   (ws + FEAT16 + W16 + CA16 + 3 * QK16 + UT16);

  cvt_kernel   <<<11008, 256, 0, stream>>>(feat, ca, w, feat16, ca16, w16);
  qkv_kernel   <<<6144, 128, 0, stream>>>(feat16, w16, q16, k16, vt16);
  rowsum_kernel<<<512,  256, 0, stream>>>(ca, s32);
  fill_kernel  <<<2048, 256, 0, stream>>>(s32, ut16);
  u_kernel     <<<2048, 128, 0, stream>>>(ca16, vt16, ut16);
  flash_kernel <<<512,  128, 0, stream>>>(q16, k16, ut16, out);
}
